// TGACN_11647951307137
// MI455X (gfx1250) — matrix operand order repaired and checked by reading, not compiled, not run on hardware
//
#include <hip/hip_runtime.h>
#include <hip/hip_bf16.h>

// ---------------- problem constants ----------------
#define Bq   32
#define Cc   48
#define Nn   512
#define Tt   24
#define OC3  16
#define NT   (Nn * Tt)   // 12288
#define CPAD 64          // channel dim padded for K of WMMA
#define TPAD 32          // time dim padded for K of WMMA

typedef __attribute__((ext_vector_type(16))) _Float16 v16h;
typedef __attribute__((ext_vector_type(8)))  _Float16 v8h;
typedef __attribute__((ext_vector_type(8)))  float    v8f;

union H16 { v16h v; v8h h[2]; };

__device__ __forceinline__ v8f wmma_f16(v16h a, v16h b, v8f c) {
  // (neg_a, A, neg_b, B, c_mod, C, reuse_a, reuse_b)
  return __builtin_amdgcn_wmma_f32_16x16x32_f16(false, a, false, b, (short)0, c,
                                                false, false);
}

// Load 16x32 f16 A fragment: row-major A, row stride lda (halves).
// lane m = L%16 is the row; halves j=0..15 -> K = k0 + hi*16 + j
// (contiguous order, matching load_b).
__device__ __forceinline__ v16h load_a(const _Float16* __restrict__ A, int row,
                                       int lda, int k0, int m, int hi) {
  const _Float16* p = A + (size_t)row * lda + k0 + hi * 16;
  H16 f;
  f.h[0] = *(const v8h*)p;
  f.h[1] = *(const v8h*)(p + 8);
  return f.v;
}

// Load 32x16 f16 B fragment from "Bt" memory holding B transposed:
// Bt[col][k], k contiguous with stride ldb. lane col = L%16,
// halves j -> K = k0 + hi*16 + j (contiguous order, matching load_a).
__device__ __forceinline__ v16h load_b(const _Float16* __restrict__ Bt, int col,
                                       int ldb, int k0, int hi) {
  return *(const v16h*)(Bt + (size_t)col * ldb + k0 + hi * 16);
}

// ---------------- K1: LayerNorm over T, emit f16 staging copies ----------------
// h16p : [B, C, N, TPAD]  (T-contiguous, padded with zeros)  -> TACN A, reductions
// h16T : [B, N*T, CPAD]   (channel-contiguous, padded zeros) -> CACN / hg A
__global__ void k_layernorm(const float* __restrict__ x,
                            const float* __restrict__ lw,
                            const float* __restrict__ lb,
                            _Float16* __restrict__ h16p,
                            _Float16* __restrict__ h16T) {
  int idx = blockIdx.x * blockDim.x + threadIdx.x;
  if (idx >= Bq * Cc * Nn) return;
  int n = idx % Nn;
  int c = (idx / Nn) % Cc;
  int b = idx / (Nn * Cc);

  const float* xr = x + (size_t)idx * Tt;
  float v[Tt];
  float mu = 0.f;
#pragma unroll
  for (int t = 0; t < Tt; ++t) { v[t] = xr[t]; mu += v[t]; }
  mu *= (1.f / Tt);
  float var = 0.f;
#pragma unroll
  for (int t = 0; t < Tt; ++t) { float d = v[t] - mu; var += d * d; }
  var *= (1.f / Tt);
  float rs = rsqrtf(var + 1e-5f);

  _Float16* hp = h16p + (size_t)idx * TPAD;
  _Float16* hT = h16T + ((size_t)b * NT + (size_t)n * Tt) * CPAD + c;
#pragma unroll
  for (int t = 0; t < Tt; ++t) {
    float hv = (v[t] - mu) * rs * lw[t] + lb[t];
    hp[t] = (_Float16)hv;
    hT[(size_t)t * CPAD] = (_Float16)hv;
  }
#pragma unroll
  for (int t = Tt; t < TPAD; ++t) hp[t] = (_Float16)0.f;
  if (c == 0) {  // zero channel-pad of h16T for this (b,n,*) group
    for (int t = 0; t < Tt; ++t)
      for (int cc = Cc; cc < CPAD; ++cc)
        h16T[((size_t)b * NT + (size_t)n * Tt + t) * CPAD + cc] = (_Float16)0.f;
  }
}

// ---------------- K2: sc[b,c,t] = sum_n h * alpha_c[n] ----------------
__global__ void k_sc(const _Float16* __restrict__ h16p,
                     const float* __restrict__ alpha_c,
                     float* __restrict__ sc) {
  int bc = blockIdx.x;            // b*Cc + c
  int t  = threadIdx.x;           // 0..31
  if (t >= Tt) return;
  const _Float16* base = h16p + (size_t)bc * Nn * TPAD;
  float s = 0.f;
  for (int n = 0; n < Nn; ++n)
    s += (float)base[(size_t)n * TPAD + t] * alpha_c[n];
  sc[(size_t)bc * Tt + t] = s;
}

// ---------------- K3: st / sg / sgG (channel reductions + tiny Gw matmul) ------
__global__ void k_stsg(const _Float16* __restrict__ h16p,
                       const float* __restrict__ alpha_t,
                       const float* __restrict__ alpha_g,
                       const float* __restrict__ Gw,
                       float* __restrict__ stb,       // [B,T,N]
                       _Float16* __restrict__ sg16,   // [B,N,TPAD]
                       _Float16* __restrict__ sgG16)  // [B,N,TPAD]
{
  int bn = blockIdx.x;
  int b = bn / Nn, n = bn % Nn;
  int t = threadIdx.x;            // 0..31
  float stv = 0.f, sgv = 0.f;
  const _Float16* base = h16p + ((size_t)b * Cc * Nn + n) * TPAD;
  for (int c = 0; c < Cc; ++c) {
    float hv = (float)base[(size_t)c * Nn * TPAD + t];
    stv += hv * alpha_t[c];
    sgv += hv * alpha_g[c];
  }
  __shared__ float sh[TPAD];
  sh[t] = (t < Tt) ? sgv : 0.f;
  __syncthreads();
  if (t < Tt) stb[((size_t)b * Tt + t) * Nn + n] = stv;
  float sgg = 0.f;
  if (t < Tt) {
    for (int u = 0; u < Tt; ++u) sgg += sh[u] * Gw[u * Tt + t];
  }
  sg16 [(size_t)bn * TPAD + t] = (_Float16)((t < Tt) ? sgv : 0.f);
  sgG16[(size_t)bn * TPAD + t] = (_Float16)((t < Tt) ? sgg : 0.f);
}

// ---------------- K4: channel attention + fold with cconv_w -> Mc16 ----------
__global__ void k_cacn_att(const float* __restrict__ sc,
                           const float* __restrict__ Wc,
                           const float* __restrict__ cconv_w,
                           _Float16* __restrict__ Mc16)  // [B,16,CPAD]
{
  int b = blockIdx.x;
  int tid = threadIdx.x;          // 64 threads
  __shared__ float s_sc [Cc * Tt];
  __shared__ float s_scw[Cc * Tt];
  __shared__ float s_log[Cc * Cc];
  for (int i = tid; i < Cc * Tt; i += 64) s_sc[i] = sc[(size_t)b * Cc * Tt + i];
  __syncthreads();
  for (int i = tid; i < Cc * Tt; i += 64) {
    int c = i / Tt, u = i % Tt;
    float s = 0.f;
    for (int t = 0; t < Tt; ++t) s += s_sc[c * Tt + t] * Wc[t * Tt + u];
    s_scw[i] = s;
  }
  __syncthreads();
  for (int i = tid; i < Cc * Cc; i += 64) {
    int c = i / Cc, d = i % Cc;
    float s = 0.f;
    for (int u = 0; u < Tt; ++u) s += s_scw[c * Tt + u] * s_sc[d * Tt + u];
    s_log[i] = s;
  }
  __syncthreads();
  if (tid < Cc) {   // softmax over d
    float mx = -3.4e38f;
    for (int d = 0; d < Cc; ++d) mx = fmaxf(mx, s_log[tid * Cc + d]);
    float sum = 0.f;
    for (int d = 0; d < Cc; ++d) {
      float e = __expf(s_log[tid * Cc + d] - mx);
      s_log[tid * Cc + d] = e; sum += e;
    }
    float inv = 1.f / sum;
    for (int d = 0; d < Cc; ++d) s_log[tid * Cc + d] *= inv;
  }
  __syncthreads();
  for (int i = tid; i < OC3 * CPAD; i += 64) {   // Mc = cconv_w @ attc
    int o = i / CPAD, ii = i % CPAD;
    float s = 0.f;
    if (ii < Cc)
      for (int c = 0; c < Cc; ++c) s += cconv_w[o * Cc + c] * s_log[c * Cc + ii];
    Mc16[(size_t)b * OC3 * CPAD + i] = (_Float16)s;
  }
}

// ---------------- K5: temporal attention -> atttb [B,TPAD,TPAD] f16 ----------
__global__ void k_tacn_att(const float* __restrict__ stb,
                           const float* __restrict__ Wt1,
                           const float* __restrict__ Wt2,
                           _Float16* __restrict__ atttb) {
  int b = blockIdx.x;
  int tid = threadIdx.x;          // 256 threads
  __shared__ float q [Tt * 10];
  __shared__ float r [10 * Tt];
  __shared__ float lg[Tt * Tt];
  const float* st = stb + (size_t)b * Tt * Nn;
  for (int i = tid; i < Tt * 10; i += 256) {
    int s = i / 10, j = i % 10;
    float acc = 0.f;
    for (int n = 0; n < Nn; ++n) acc += st[s * Nn + n] * Wt1[j * Nn + n];
    q[i] = acc;
  }
  for (int i = tid; i < 10 * Tt; i += 256) {
    int j = i / Tt, t = i % Tt;
    float acc = 0.f;
    for (int n = 0; n < Nn; ++n) acc += Wt2[j * Nn + n] * st[t * Nn + n];
    r[i] = acc;
  }
  __syncthreads();
  for (int i = tid; i < Tt * Tt; i += 256) {
    int s = i / Tt, t = i % Tt;
    float acc = 0.f;
    for (int j = 0; j < 10; ++j) acc += q[s * 10 + j] * r[j * Tt + t];
    lg[i] = acc;
  }
  __syncthreads();
  if (tid < Tt) {   // softmax over t
    float mx = -3.4e38f;
    for (int t = 0; t < Tt; ++t) mx = fmaxf(mx, lg[tid * Tt + t]);
    float sum = 0.f;
    for (int t = 0; t < Tt; ++t) {
      float e = __expf(lg[tid * Tt + t] - mx);
      lg[tid * Tt + t] = e; sum += e;
    }
    float inv = 1.f / sum;
    for (int t = 0; t < Tt; ++t) lg[tid * Tt + t] *= inv;
  }
  __syncthreads();
  for (int i = tid; i < TPAD * TPAD; i += 256) {
    int s = i / TPAD, t = i % TPAD;
    atttb[(size_t)b * TPAD * TPAD + i] =
        (_Float16)((s < Tt && t < Tt) ? lg[s * Tt + t] : 0.f);
  }
}

// ---------------- K6: graph logits GEMM (WMMA): sgG @ sg^T -> glog [B,N,N] ----
__global__ void k_glogits(const _Float16* __restrict__ sgG16,
                          const _Float16* __restrict__ sg16,
                          float* __restrict__ glog) {
  int bid = blockIdx.x;                 // b*1024 + mt*32 + ct
  int ct = bid & 31, mt = (bid >> 5) & 31, b = bid >> 10;
  int lane = threadIdx.x, m = lane & 15, hi = lane >> 4;
  const _Float16* A  = sgG16 + (size_t)b * Nn * TPAD;
  const _Float16* Bt = sg16  + (size_t)b * Nn * TPAD;
  v8f acc = {};
  v16h a = load_a(A, mt * 16 + m, TPAD, 0, m, hi);
  v16h bb = load_b(Bt, ct * 16 + m, TPAD, 0, hi);
  acc = wmma_f16(a, bb, acc);
  float* D = glog + (size_t)b * Nn * Nn;
  int col = ct * 16 + m;
#pragma unroll
  for (int rr = 0; rr < 8; ++rr) {
    int row = mt * 16 + rr + 8 * hi;
    D[(size_t)row * Nn + col] = acc[rr];
  }
}

// ---------------- K7: softmax over m, then * adj -> Ag f16 [B,N,N] -----------
__global__ void k_attg_softmax(const float* __restrict__ glog,
                               const float* __restrict__ adj,
                               _Float16* __restrict__ Ag) {
  int bn = blockIdx.x;
  int n = bn % Nn;
  int tid = threadIdx.x;          // 256
  const float* row = glog + (size_t)bn * Nn;
  __shared__ float red[256];
  float mx = -3.4e38f;
  for (int m = tid; m < Nn; m += 256) mx = fmaxf(mx, row[m]);
  red[tid] = mx; __syncthreads();
  for (int s = 128; s > 0; s >>= 1) {
    if (tid < s) red[tid] = fmaxf(red[tid], red[tid + s]);
    __syncthreads();
  }
  mx = red[0]; __syncthreads();
  float sum = 0.f;
  for (int m = tid; m < Nn; m += 256) sum += __expf(row[m] - mx);
  red[tid] = sum; __syncthreads();
  for (int s = 128; s > 0; s >>= 1) {
    if (tid < s) red[tid] += red[tid + s];
    __syncthreads();
  }
  float inv = 1.f / red[0];
  for (int m = tid; m < Nn; m += 256) {
    float a = __expf(row[m] - mx) * inv * adj[(size_t)n * Nn + m];
    Ag[(size_t)bn * Nn + m] = (_Float16)a;
  }
}

// ---------------- K8: hg = Wg @ h (WMMA), store transposed hgT [B,16,TPAD,N] --
__global__ void k_hg(const _Float16* __restrict__ h16T,
                     const _Float16* __restrict__ Wg16,   // [16,CPAD] Bt layout
                     _Float16* __restrict__ hgT) {
  int bid = blockIdx.x;
  int mt = bid % (NT / 16);
  int b  = bid / (NT / 16);
  int lane = threadIdx.x, m = lane & 15, hi = lane >> 4;
  const _Float16* A = h16T + (size_t)b * NT * CPAD;
  v8f acc = {};
#pragma unroll
  for (int k0 = 0; k0 < CPAD; k0 += 32) {
    v16h a = load_a(A, mt * 16 + m, CPAD, k0, m, hi);
    v16h bb = load_b(Wg16, m, CPAD, k0, hi);   // col = o = m
    acc = wmma_f16(a, bb, acc);
  }
  int o = m;
#pragma unroll
  for (int rr = 0; rr < 8; ++rr) {
    int ntrow = mt * 16 + rr + 8 * hi;         // (i,t) flat index
    int ni = ntrow / Tt, tt = ntrow % Tt;
    hgT[(((size_t)b * OC3 + o) * TPAD + tt) * Nn + ni] = (_Float16)acc[rr];
  }
}

// ---------------- K9: GACN main GEMM (WMMA + LDS-staged B + A-in-regs) -------
// Block = 4 waves, handles (b, group of 4 m-tiles) for ALL 16 output channels.
// A panel (16 rows x 512 K) lives in 128 VGPRs per wave (loaded once).
// B panel (32 x 512, 32KB) is staged in LDS once per o and read via ds_load.
#define GACN_WAVES 4
__global__ void __launch_bounds__(32 * GACN_WAVES)
k_gacn(const _Float16* __restrict__ Ag,
       const _Float16* __restrict__ hgT,
       float* __restrict__ g_out) {      // [B,16,N,T]
  __shared__ _Float16 sB[TPAD * Nn];     // 32 KB B panel for current o
  int bid = blockIdx.x;                  // b*8 + mtg
  int mtg = bid & 7;
  int b   = bid >> 3;
  int tid  = threadIdx.x;
  int wave = tid >> 5;
  int lane = tid & 31;
  int m = lane & 15, hi = lane >> 4;
  int mt = mtg * GACN_WAVES + wave;

  // ---- A tile for this wave: 16 K-step fragments held in registers ----
  const _Float16* pa =
      Ag + (size_t)b * Nn * Nn + (size_t)(mt * 16 + m) * Nn + hi * 16;
  H16 af[16];
#pragma unroll
  for (int ks = 0; ks < 16; ++ks) {
    af[ks].h[0] = *(const v8h*)(pa + ks * 32);
    af[ks].h[1] = *(const v8h*)(pa + ks * 32 + 8);
  }

  const size_t panelElems = (size_t)TPAD * Nn;   // 16384 halves = 32 KB
  for (int o = 0; o < OC3; ++o) {
    const _Float16* Bp = hgT + ((size_t)b * OC3 + o) * panelElems;
    __syncthreads();   // previous iteration's LDS reads complete
    // cooperative global -> LDS stage: 128 threads x 8 x 32B chunks
#pragma unroll
    for (int j = 0; j < 8; ++j) {
      int idx16 = j * (32 * GACN_WAVES) + tid;   // 0..1023 chunks of 16 halves
      *(v16h*)(sB + idx16 * 16) = *(const v16h*)(Bp + idx16 * 16);
    }
    __syncthreads();
    if (o + 1 < OC3) {  // prefetch next o's panel while we compute
      __builtin_prefetch(Bp + panelElems + (size_t)tid * 128, 0, 1);
    }
    v8f acc0 = {}, acc1 = {};
    const _Float16* pb0 = sB + (size_t)m * Nn + hi * 16;          // cols 0..15
    const _Float16* pb1 = sB + (size_t)(16 + m) * Nn + hi * 16;   // cols 16..31
#pragma unroll
    for (int ks = 0; ks < 16; ++ks) {
      v16h b0 = *(const v16h*)(pb0 + ks * 32);
      v16h b1 = *(const v16h*)(pb1 + ks * 32);
      acc0 = wmma_f16(af[ks].v, b0, acc0);
      acc1 = wmma_f16(af[ks].v, b1, acc1);
    }
#pragma unroll
    for (int rr = 0; rr < 8; ++rr) {
      int n = mt * 16 + rr + 8 * hi;
      float* dst = g_out + (((size_t)b * OC3 + o) * Nn + n) * Tt;
      dst[m] = acc0[rr];                 // t = m (< 16)
      if (16 + m < Tt) dst[16 + m] = acc1[rr];
    }
  }
}

// ---------------- K10: CACN GEMM (WMMA): out_c^T = h16T @ Mc^T --------------
__global__ void k_cacn(const _Float16* __restrict__ h16T,
                       const _Float16* __restrict__ Mc16,
                       float* __restrict__ c_out) {      // [B,16,NT]
  int bid = blockIdx.x;
  int mt = bid % (NT / 16);
  int b  = bid / (NT / 16);
  int lane = threadIdx.x, m = lane & 15, hi = lane >> 4;
  const _Float16* A  = h16T + (size_t)b * NT * CPAD;
  const _Float16* Bt = Mc16 + (size_t)b * OC3 * CPAD;
  v8f acc = {};
#pragma unroll
  for (int k0 = 0; k0 < CPAD; k0 += 32) {
    v16h a = load_a(A, mt * 16 + m, CPAD, k0, m, hi);
    v16h bb = load_b(Bt, m, CPAD, k0, hi);     // col = o = m
    acc = wmma_f16(a, bb, acc);
  }
  int o = m;
#pragma unroll
  for (int rr = 0; rr < 8; ++rr) {
    int ntrow = mt * 16 + rr + 8 * hi;
    c_out[((size_t)b * OC3 + o) * NT + ntrow] = acc[rr];
  }
}

// ---------------- K11: TACN temporal mix GEMM (WMMA): t1 = h @ attt^T --------
__global__ void k_tacn(const _Float16* __restrict__ h16p,
                       const _Float16* __restrict__ atttb,
                       float* __restrict__ t1) {         // [B,48,N,T]
  int bid = blockIdx.x;
  int mt = bid % (Cc * Nn / 16);
  int b  = bid / (Cc * Nn / 16);
  int lane = threadIdx.x, m = lane & 15, hi = lane >> 4;
  const _Float16* A  = h16p  + (size_t)b * Cc * Nn * TPAD;
  const _Float16* Bt = atttb + (size_t)b * TPAD * TPAD;
  v16h a  = load_a(A, mt * 16 + m, TPAD, 0, m, hi);
  v16h b0 = load_b(Bt, m,      TPAD, 0, hi);   // cols s=0..15
  v16h b1 = load_b(Bt, 16 + m, TPAD, 0, hi);   // cols s=16..31
  v8f acc0 = {}, acc1 = {};
  acc0 = wmma_f16(a, b0, acc0);
  acc1 = wmma_f16(a, b1, acc1);
#pragma unroll
  for (int rr = 0; rr < 8; ++rr) {
    int row = mt * 16 + rr + 8 * hi;           // c*Nn + n
    float* dst = t1 + ((size_t)b * Cc * Nn + row) * Tt;
    dst[m] = acc0[rr];
    if (16 + m < Tt) dst[16 + m] = acc1[rr];
  }
}

// ---------------- K12: causal dilated conv [1,2] -----------------------------
__global__ void k_causal_conv(const float* __restrict__ in,
                              const float* __restrict__ w,
                              const float* __restrict__ bias,
                              float* __restrict__ out, int Cin, int d) {
  int idx = blockIdx.x * blockDim.x + threadIdx.x;
  if (idx >= Bq * OC3 * Nn) return;
  int n = idx % Nn;
  int o = (idx / Nn) % OC3;
  int b = idx / (Nn * OC3);
  float acc[Tt];
  float bb = bias[o];
#pragma unroll
  for (int t = 0; t < Tt; ++t) acc[t] = bb;
  const float* ib = in + (size_t)b * Cin * Nn * Tt + (size_t)n * Tt;
  for (int c = 0; c < Cin; ++c) {
    const float* row = ib + (size_t)c * Nn * Tt;
    float w0 = w[(o * Cin + c) * 2 + 0];   // taps x[t-d]
    float w1 = w[(o * Cin + c) * 2 + 1];   // taps x[t]
#pragma unroll
    for (int t = 0; t < Tt; ++t) {
      float cur  = row[t];
      float prev = (t >= d) ? row[t - d] : 0.f;
      acc[t] += w1 * cur + w0 * prev;
    }
  }
  float* ob = out + ((size_t)b * OC3 + o) * Nn * Tt + (size_t)n * Tt;
#pragma unroll
  for (int t = 0; t < Tt; ++t) ob[t] = acc[t];
}

// ---------------- K_w: Wg -> f16 Bt layout [16,CPAD] -------------------------
__global__ void k_wg16(const float* __restrict__ Wg, _Float16* __restrict__ Wg16) {
  int i = blockIdx.x * blockDim.x + threadIdx.x;
  if (i < OC3 * CPAD) {
    int o = i / CPAD, c = i % CPAD;
    Wg16[i] = (_Float16)((c < Cc) ? Wg[o * Cc + c] : 0.f);
  }
}

// ---------------- K15: concat + residual + relu ------------------------------
__global__ void k_final(const float* __restrict__ x,
                        const float* __restrict__ res_w,
                        const float* __restrict__ res_b,
                        const float* __restrict__ cconv_b,
                        const float* __restrict__ c_out,
                        const float* __restrict__ t_out,
                        const float* __restrict__ g_out,
                        float* __restrict__ out) {
  __shared__ float sw[Cc * Cc];
  int tid = threadIdx.x;
  for (int i = tid; i < Cc * Cc; i += blockDim.x) sw[i] = res_w[i];
  __syncthreads();
  int idx = blockIdx.x * blockDim.x + tid;
  if (idx >= Bq * Nn * Tt) return;
  int t = idx % Tt;
  int n = (idx / Tt) % Nn;
  int b = idx / (Nn * Tt);
  float xv[Cc];
  size_t base = (size_t)b * Cc * Nn * Tt + (size_t)n * Tt + t;
#pragma unroll
  for (int c = 0; c < Cc; ++c) xv[c] = x[base + (size_t)c * Nn * Tt];
  size_t nt = (size_t)n * Tt + t;
  for (int o = 0; o < Cc; ++o) {
    float r = res_b[o];
    for (int c = 0; c < Cc; ++c) r += sw[o * Cc + c] * xv[c];
    float m;
    if (o < OC3)
      m = c_out[((size_t)b * OC3 + o) * NT + nt] + cconv_b[o];
    else if (o < 2 * OC3)
      m = t_out[(((size_t)b * OC3 + (o - OC3)) * Nn + n) * Tt + t];
    else
      m = g_out[(((size_t)b * OC3 + (o - 2 * OC3)) * Nn + n) * Tt + t];
    out[((size_t)b * Cc + o) * Nn * Tt + nt] = fmaxf(m + r, 0.f);
  }
}

// ============================ launch =========================================
extern "C" void kernel_launch(void* const* d_in, const int* in_sizes, int n_in,
                              void* d_out, int out_size, void* d_ws, size_t ws_size,
                              hipStream_t stream) {
  const float* x        = (const float*)d_in[0];
  const float* adj      = (const float*)d_in[1];
  const float* ln_w     = (const float*)d_in[2];
  const float* ln_b     = (const float*)d_in[3];
  const float* res_w    = (const float*)d_in[4];
  const float* res_b    = (const float*)d_in[5];
  const float* Wc       = (const float*)d_in[6];
  const float* alpha_c  = (const float*)d_in[7];
  const float* cconv_w  = (const float*)d_in[8];
  const float* cconv_b  = (const float*)d_in[9];
  const float* Wt1      = (const float*)d_in[10];
  const float* Wt2      = (const float*)d_in[11];
  const float* alpha_t  = (const float*)d_in[12];
  const float* tc0_w    = (const float*)d_in[13];
  const float* tc0_b    = (const float*)d_in[14];
  const float* tc1_w    = (const float*)d_in[15];
  const float* tc1_b    = (const float*)d_in[16];
  const float* tc2_w    = (const float*)d_in[17];
  const float* tc2_b    = (const float*)d_in[18];
  const float* Gw       = (const float*)d_in[19];
  const float* alpha_g  = (const float*)d_in[20];
  const float* Wg       = (const float*)d_in[21];
  float* out = (float*)d_out;

  // ---- workspace carve-up (256B aligned) ----
  char* ws = (char*)d_ws;
  size_t off = 0;
  auto carve = [&](size_t bytes) {
    size_t o = off;
    off += (bytes + 255) & ~(size_t)255;
    return (void*)(ws + o);
  };
  _Float16* h16T  = (_Float16*)carve((size_t)Bq * NT * CPAD * 2);
  _Float16* h16p  = (_Float16*)carve((size_t)Bq * Cc * Nn * TPAD * 2);
  float*    sc    = (float*)   carve((size_t)Bq * Cc * Tt * 4);
  float*    stb   = (float*)   carve((size_t)Bq * Tt * Nn * 4);
  _Float16* sg16  = (_Float16*)carve((size_t)Bq * Nn * TPAD * 2);
  _Float16* sgG16 = (_Float16*)carve((size_t)Bq * Nn * TPAD * 2);
  _Float16* Mc16  = (_Float16*)carve((size_t)Bq * OC3 * CPAD * 2);
  _Float16* atttb = (_Float16*)carve((size_t)Bq * TPAD * TPAD * 2);
  _Float16* Wg16  = (_Float16*)carve((size_t)OC3 * CPAD * 2);
  float*    glog  = (float*)   carve((size_t)Bq * Nn * Nn * 4);
  _Float16* Ag    = (_Float16*)carve((size_t)Bq * Nn * Nn * 2);
  _Float16* hgT   = (_Float16*)carve((size_t)Bq * OC3 * TPAD * Nn * 2);
  float*    c_out = (float*)   carve((size_t)Bq * OC3 * NT * 4);
  float*    t1    = (float*)   carve((size_t)Bq * Cc * Nn * Tt * 4);
  float*    ta    = (float*)   carve((size_t)Bq * OC3 * Nn * Tt * 4);
  float*    tb    = (float*)   carve((size_t)Bq * OC3 * Nn * Tt * 4);
  float*    g_out = (float*)   carve((size_t)Bq * OC3 * Nn * Tt * 4);

  // ---- pipeline (single stream -> ordering == dependencies) ----
  k_wg16<<<(OC3 * CPAD + 255) / 256, 256, 0, stream>>>(Wg, Wg16);
  k_layernorm<<<(Bq * Cc * Nn + 255) / 256, 256, 0, stream>>>(x, ln_w, ln_b,
                                                              h16p, h16T);
  k_sc<<<Bq * Cc, 32, 0, stream>>>(h16p, alpha_c, sc);
  k_stsg<<<Bq * Nn, 32, 0, stream>>>(h16p, alpha_t, alpha_g, Gw, stb, sg16, sgG16);
  k_cacn_att<<<Bq, 64, 0, stream>>>(sc, Wc, cconv_w, Mc16);
  k_tacn_att<<<Bq, 256, 0, stream>>>(stb, Wt1, Wt2, atttb);

  // GACN attention logits + softmax*adj
  k_glogits<<<Bq * 32 * 32, 32, 0, stream>>>(sgG16, sg16, glog);
  k_attg_softmax<<<Bq * Nn, 256, 0, stream>>>(glog, adj, Ag);

  // hg = Wg @ h, stored transposed + zero-padded rows
  hipMemsetAsync(hgT, 0, (size_t)Bq * OC3 * TPAD * Nn * 2, stream);
  k_hg<<<Bq * (NT / 16), 32, 0, stream>>>(h16T, Wg16, hgT);

  // heavy GEMMs
  k_gacn<<<Bq * 8, 32 * GACN_WAVES, 0, stream>>>(Ag, hgT, g_out);
  k_cacn<<<Bq * (NT / 16), 32, 0, stream>>>(h16T, Mc16, c_out);
  k_tacn<<<Bq * (Cc * Nn / 16), 32, 0, stream>>>(h16p, atttb, t1);

  // dilated causal conv chain (48 -> 16 -> 16 -> 16)
  int convThreads = Bq * OC3 * Nn;
  k_causal_conv<<<(convThreads + 255) / 256, 256, 0, stream>>>(t1, tc0_w, tc0_b,
                                                               ta, Cc, 1);
  k_causal_conv<<<(convThreads + 255) / 256, 256, 0, stream>>>(ta, tc1_w, tc1_b,
                                                               tb, OC3, 2);
  k_causal_conv<<<(convThreads + 255) / 256, 256, 0, stream>>>(tb, tc2_w, tc2_b,
                                                               ta, OC3, 4);

  // concat + residual + relu
  k_final<<<(Bq * Nn * Tt + 255) / 256, 256, 0, stream>>>(
      x, res_w, res_b, cconv_b, c_out, ta, g_out, out);
}
